// QuerySelector_4853313045191
// MI455X (gfx1250) — matrix operand order repaired and checked by reading, not compiled, not run on hardware
//
#include <hip/hip_runtime.h>
#include <hip/hip_bf16.h>
#include <math.h>

#define B_   8
#define LQ   2744
#define LK   4096
#define NQ   4096
#define D_   512
#define QTILES 172   // ceil(2744/16)

typedef __attribute__((ext_vector_type(16))) __bf16 v16bf;
typedef __attribute__((ext_vector_type(8)))  float  v8f;
typedef __attribute__((ext_vector_type(4)))  int    v4i;

// Native converts -> v_cvt_pk_bf16_f32
__device__ __forceinline__ __bf16 f2bf(float f) { return (__bf16)f; }
__device__ __forceinline__ unsigned short f2bf_bits(float f) {
    return __builtin_bit_cast(unsigned short, (__bf16)f);
}

// ---- CDNA5 async-to-LDS path (probe-verified: takes generic v4i* args) ----
#if defined(__AMDGCN__) && __has_builtin(__builtin_amdgcn_global_load_async_to_lds_b128)
#define ASYNC_LDS_COPY 1
#endif

__device__ __forceinline__ void wait_async_zero() {
#if defined(__AMDGCN__)
#if __has_builtin(__builtin_amdgcn_s_wait_asynccnt)
    __builtin_amdgcn_s_wait_asynccnt(0);
#else
    asm volatile("s_wait_asynccnt 0x0" ::: "memory");
#endif
#endif
}

// ---------------------------------------------------------------------------
// Kernel 1: K_reduce[b,d] = mean of top-2744 of keys[b,:,d]
// ---------------------------------------------------------------------------
__global__ __launch_bounds__(256) void kreduce_kernel(const float* __restrict__ keys,
                                                      float* __restrict__ kred) {
    const int d = blockIdx.x, b = blockIdx.y, tid = threadIdx.x;
    __shared__ float buf[LK];
    __shared__ float red[256];
    const size_t base = (size_t)b * LK * D_ + d;
    for (int i = tid; i < LK; i += 256) buf[i] = keys[base + (size_t)i * D_];
    __syncthreads();
    for (int k = 2; k <= LK; k <<= 1) {
        for (int j = k >> 1; j > 0; j >>= 1) {
            for (int i = tid; i < LK; i += 256) {
                int ixj = i ^ j;
                if (ixj > i) {
                    bool up = ((i & k) == 0);
                    float a = buf[i], c = buf[ixj];
                    if ((a > c) == up) { buf[i] = c; buf[ixj] = a; }
                }
            }
            __syncthreads();
        }
    }
    float s = 0.f;
    for (int i = (LK - LQ) + tid; i < LK; i += 256) s += buf[i];
    red[tid] = s; __syncthreads();
    for (int off = 128; off > 0; off >>= 1) {
        if (tid < off) red[tid] += red[tid + off];
        __syncthreads();
    }
    if (tid == 0) kred[b * D_ + d] = red[0] * (1.0f / (float)LQ);
}

// ---------------------------------------------------------------------------
// Kernel 2: sqk[b,q] = dot(K_reduce[b,:], queries[b,q,:]); wave per query
// ---------------------------------------------------------------------------
__global__ __launch_bounds__(256) void sqk_kernel(const float* __restrict__ q,
                                                  const float* __restrict__ kred,
                                                  float* __restrict__ sqk) {
    const int b = blockIdx.y;
    const int wave = threadIdx.x >> 5, lane = threadIdx.x & 31;
    const int qrow = blockIdx.x * 8 + wave;
    const float* qp = q + ((size_t)b * NQ + qrow) * D_;
    const float* kp = kred + b * D_;
    float s = 0.f;
    for (int i = lane; i < D_; i += 32) s += qp[i] * kp[i];
    for (int off = 16; off > 0; off >>= 1) s += __shfl_down(s, off, 32);
    if (lane == 0) sqk[b * NQ + qrow] = s;
}

// ---------------------------------------------------------------------------
// Kernel 3: per-batch top-2744 indices of sqk (desc value, asc index on ties)
// ---------------------------------------------------------------------------
__global__ __launch_bounds__(256) void topq_kernel(const float* __restrict__ sqk,
                                                   int* __restrict__ indices) {
    const int b = blockIdx.x, tid = threadIdx.x;
    __shared__ unsigned long long keybuf[NQ];
    for (int i = tid; i < NQ; i += 256) {
        unsigned u = __float_as_uint(sqk[b * NQ + i]);
        unsigned m = (u & 0x80000000u) ? ~u : (u | 0x80000000u);
        unsigned k32 = ~m;
        keybuf[i] = ((unsigned long long)k32 << 32) | (unsigned)i;
    }
    __syncthreads();
    for (int k = 2; k <= NQ; k <<= 1) {
        for (int j = k >> 1; j > 0; j >>= 1) {
            for (int i = tid; i < NQ; i += 256) {
                int ixj = i ^ j;
                if (ixj > i) {
                    bool up = ((i & k) == 0);
                    unsigned long long a = keybuf[i], c = keybuf[ixj];
                    if ((a > c) == up) { keybuf[i] = c; keybuf[ixj] = a; }
                }
            }
            __syncthreads();
        }
    }
    for (int i = tid; i < LQ; i += 256)
        indices[b * LQ + i] = (int)(keybuf[i] & 0xFFFFFFFFu);
}

// ---------------------------------------------------------------------------
// Kernel 4/5: column means of V; broadcast fill of the output
// ---------------------------------------------------------------------------
__global__ __launch_bounds__(256) void meanv_kernel(const float* __restrict__ values,
                                                    float* __restrict__ meanv) {
    const int b = blockIdx.y;
    const int d = blockIdx.x * 256 + threadIdx.x;
    const float* vp = values + (size_t)b * LK * D_ + d;
    float s = 0.f;
    for (int r = 0; r < LK; ++r) s += vp[(size_t)r * D_];
    meanv[b * D_ + d] = s * (1.0f / (float)LK);
}

__global__ __launch_bounds__(256) void fill_kernel(const float* __restrict__ meanv,
                                                   float* __restrict__ out) {
    size_t idx = (size_t)blockIdx.x * 256 + threadIdx.x;
    int d = (int)(idx & (D_ - 1));
    int b = (int)(idx >> 21);
    out[idx] = meanv[b * D_ + d];
}

// ---------------------------------------------------------------------------
// Kernel 6: bf16 WMMA flash attention over the selected queries.
//  Q stage : gathered rows copied global->LDS via ASYNC DMA (ASYNCcnt), then
//            one cvt pass to bf16 (Vs slab reused as the raw landing buffer)
//  S stage : wave w -> 16x16 S tile vs keys [kc+16w,+16), K=512 (16 WMMA)
//  softmax : parallel (16 threads/row, shfl_xor reductions)
//  PV stage: V slab (32 keys x 512) staged transposed in LDS; B frags are
//            contiguous ds_load_b128. wave w owns cols [64w,64w+64): 4x4 WMMA
// ---------------------------------------------------------------------------
__global__ __launch_bounds__(256) void attn_kernel(const float* __restrict__ queries,
                                                   const float* __restrict__ keys,
                                                   const float* __restrict__ values,
                                                   const int* __restrict__ indices,
                                                   float* __restrict__ out) {
    const int b = blockIdx.y;
    const int qtile = blockIdx.x;
    const int tid = threadIdx.x;
    const int wave = tid >> 5;
    const int lane = tid & 31;
    const int hi = (lane >> 4) & 1;
    const int lrow = lane & 15;

    __shared__ unsigned short Qs[16 * D_];    // 16 KB bf16 Q tile
    __shared__ float Ss[16 * 128];            // 8 KB raw scores
    __shared__ unsigned short Ps[16 * 128];   // 4 KB probs bf16
    __shared__ unsigned short Vs[D_ * 32];    // 32 KB: V slab (transposed) in the
                                              //        kc loop; raw fp32 Q at init
    __shared__ int qidx[16];
    __shared__ float rowm[16], rowl[16], rowscale[16];

    if (tid < 16) {
        int qg = qtile * 16 + tid;
        qidx[tid] = indices[b * LQ + (qg < LQ ? qg : LQ - 1)];
        rowm[tid] = -3.0e38f;
        rowl[tid] = 0.f;
    }
    __syncthreads();

    // ---- Q tile gather: async DMA to LDS, then convert to bf16 -----------
#if defined(ASYNC_LDS_COPY)
    {
        float* Qraw = (float*)Vs;             // 16*512*4B == 32 KB == sizeof(Vs)
        #pragma unroll
        for (int j = 0; j < 8; ++j) {
            int i = tid + j * 256;            // b128 index: 2048 total
            int r = i >> 7, c = (i & 127) << 2;
            const float* gsrc = queries + ((size_t)b * NQ + qidx[r]) * D_ + c;
            __builtin_amdgcn_global_load_async_to_lds_b128(
                (v4i*)gsrc, (v4i*)(Qraw + (size_t)i * 4), 0, 0);
        }
        wait_async_zero();
        __syncthreads();
        for (int i = tid; i < 16 * D_; i += 256) Qs[i] = f2bf_bits(Qraw[i]);
    }
#else
    for (int i = tid; i < 16 * D_; i += 256) {
        int r = i >> 9, d = i & (D_ - 1);
        Qs[i] = f2bf_bits(queries[((size_t)b * NQ + qidx[r]) * D_ + d]);
    }
#endif
    __syncthreads();

    v8f acc[4];
    #pragma unroll
    for (int t = 0; t < 4; ++t)
        #pragma unroll
        for (int v = 0; v < 8; ++v) acc[t][v] = 0.f;

    const float scale = 0.04419417382415922f; // 1/sqrt(512)

    for (int kc = 0; kc < LK; kc += 128) {
        // ---- S tile: this wave's 16 keys -------------------------------
        v8f s;
        #pragma unroll
        for (int v = 0; v < 8; ++v) s[v] = 0.f;
        const int keyrow = kc + wave * 16 + lrow;
        const float* kptr = keys + ((size_t)b * LK + keyrow) * D_;
        if (kc + 128 < LK)
            __builtin_prefetch(kptr + (size_t)128 * D_, 0, 0); // global_prefetch_b8
        #pragma unroll 2
        for (int dc = 0; dc < D_; dc += 32) {
            v16bf af;
            {
                const int rbase = lrow * D_ + dc;
                #pragma unroll
                for (int e = 0; e < 16; ++e) {
                    int kk = hi * 16 + e;
                    af[e] = __builtin_bit_cast(__bf16, Qs[rbase + kk]);
                }
            }
            v16bf bfr;
            {
                const float4* kp4 = (const float4*)(kptr + dc + hi * 16);
                float4 x0 = kp4[0], x1 = kp4[1], x2 = kp4[2], x3 = kp4[3];
                bfr[0] = f2bf(x0.x);  bfr[1] = f2bf(x0.y);
                bfr[2] = f2bf(x0.z);  bfr[3] = f2bf(x0.w);
                bfr[4] = f2bf(x1.x);  bfr[5] = f2bf(x1.y);
                bfr[6] = f2bf(x1.z);  bfr[7] = f2bf(x1.w);
                bfr[8] = f2bf(x2.x);  bfr[9] = f2bf(x2.y);
                bfr[10] = f2bf(x2.z); bfr[11] = f2bf(x2.w);
                bfr[12] = f2bf(x3.x); bfr[13] = f2bf(x3.y);
                bfr[14] = f2bf(x3.z); bfr[15] = f2bf(x3.w);
            }
            s = __builtin_amdgcn_wmma_f32_16x16x32_bf16(false, af, false, bfr,
                                                        (short)0, s, false, false);
        }
        #pragma unroll
        for (int v = 0; v < 8; ++v) {
            int m = v + hi * 8;
            Ss[m * 128 + wave * 16 + lrow] = s[v] * scale;
        }
        __syncthreads();

        // ---- parallel online softmax: 16 threads per row ----------------
        {
            const int r = tid >> 4, sub = tid & 15;
            const float mold = rowm[r];
            float mx = mold;
            for (int j = sub; j < 128; j += 16) mx = fmaxf(mx, Ss[r * 128 + j]);
            #pragma unroll
            for (int off = 8; off > 0; off >>= 1)
                mx = fmaxf(mx, __shfl_xor(mx, off, 16));
            float sum = 0.f;
            for (int j = sub; j < 128; j += 16) {
                float p = __expf(Ss[r * 128 + j] - mx);
                Ps[r * 128 + j] = f2bf_bits(p);
                sum += p;
            }
            #pragma unroll
            for (int off = 8; off > 0; off >>= 1)
                sum += __shfl_xor(sum, off, 16);
            if (sub == 0) {
                float c = __expf(mold - mx);
                rowl[r] = rowl[r] * c + sum;
                rowm[r] = mx;
                rowscale[r] = c;
            }
        }
        __syncthreads();

        // ---- rescale accumulators ---------------------------------------
        float sc[8];
        #pragma unroll
        for (int v = 0; v < 8; ++v) sc[v] = rowscale[v + hi * 8];
        #pragma unroll
        for (int t = 0; t < 4; ++t)
            #pragma unroll
            for (int v = 0; v < 8; ++v) acc[t][v] *= sc[v];

        // ---- P @ V with transposed LDS staging of V ---------------------
        for (int kb = 0; kb < 128; kb += 32) {
            for (int i = tid; i < (32 * D_) / 4; i += 256) {
                int r = i >> 7;                 // key row 0..31
                int c4 = (i & 127) << 2;        // d col base (x4)
                const float4 x = *(const float4*)(values +
                    ((size_t)b * LK + kc + kb + r) * D_ + c4);
                Vs[(c4 + 0) * 32 + r] = f2bf_bits(x.x);
                Vs[(c4 + 1) * 32 + r] = f2bf_bits(x.y);
                Vs[(c4 + 2) * 32 + r] = f2bf_bits(x.z);
                Vs[(c4 + 3) * 32 + r] = f2bf_bits(x.w);
            }
            __syncthreads();
            #pragma unroll
            for (int t = 0; t < 4; ++t) {
                const int dcol = wave * 64 + t * 16 + lrow;
                v16bf af;
                {
                    const int rbase = lrow * 128 + kb;
                    #pragma unroll
                    for (int e = 0; e < 16; ++e) {
                        int kk = hi * 16 + e;
                        af[e] = __builtin_bit_cast(__bf16, Ps[rbase + kk]);
                    }
                }
                v16bf bv;   // 16 consecutive ushorts -> ds_load_b128
                {
                    const unsigned short* vp = &Vs[dcol * 32 + hi * 16];
                    #pragma unroll
                    for (int e = 0; e < 16; ++e)
                        bv[e] = __builtin_bit_cast(__bf16, vp[e]);
                }
                acc[t] = __builtin_amdgcn_wmma_f32_16x16x32_bf16(false, af, false, bv,
                                                                 (short)0, acc[t],
                                                                 false, false);
            }
            __syncthreads();
        }
    }

    // ---- normalize and scatter rows to out ------------------------------
    if (tid < 16) rowscale[tid] = 1.0f / rowl[tid];
    __syncthreads();
    float inv[8];
    #pragma unroll
    for (int v = 0; v < 8; ++v) inv[v] = rowscale[v + hi * 8];
    #pragma unroll
    for (int t = 0; t < 4; ++t) {
        const int dcol = wave * 64 + t * 16 + lrow;
        #pragma unroll
        for (int v = 0; v < 8; ++v) {
            int m = v + hi * 8;
            int qg = qtile * 16 + m;
            if (qg < LQ)
                out[((size_t)b * NQ + qidx[m]) * D_ + dcol] = acc[t][v] * inv[v];
        }
    }
}

// ---------------------------------------------------------------------------
extern "C" void kernel_launch(void* const* d_in, const int* in_sizes, int n_in,
                              void* d_out, int out_size, void* d_ws, size_t ws_size,
                              hipStream_t stream) {
    (void)in_sizes; (void)n_in; (void)out_size; (void)ws_size;
    const float* queries = (const float*)d_in[0];
    const float* keys    = (const float*)d_in[1];
    const float* values  = (const float*)d_in[2];
    float* out = (float*)d_out;

    float* ws    = (float*)d_ws;
    float* kred  = ws;                       // B*D   = 4096 floats
    float* sqk   = ws + 4096;                // B*NQ  = 32768 floats
    float* meanv = ws + 4096 + 32768;        // B*D   = 4096 floats
    int*   idx   = (int*)(ws + 4096 + 32768 + 4096);  // B*LQ = 21952 ints

    kreduce_kernel<<<dim3(D_, B_), 256, 0, stream>>>(keys, kred);
    sqk_kernel   <<<dim3(NQ / 8, B_), 256, 0, stream>>>(queries, kred, sqk);
    topq_kernel  <<<B_, 256, 0, stream>>>(sqk, idx);
    meanv_kernel <<<dim3(D_ / 256, B_), 256, 0, stream>>>(values, meanv);
    fill_kernel  <<<(unsigned)((size_t)B_ * NQ * D_ / 256), 256, 0, stream>>>(meanv, out);
    attn_kernel  <<<dim3(QTILES, B_), 256, 0, stream>>>(queries, keys, values, idx, out);
}
